// HeteroGATLayer_56384330662059
// MI455X (gfx1250) — matrix operand order repaired and checked by reading, not compiled, not run on hardware
//
#include <hip/hip_runtime.h>

typedef __attribute__((ext_vector_type(16))) _Float16 v16h;
typedef __attribute__((ext_vector_type(8)))  float    v8f;

#define N_TWEET 100000
#define N_USER  50000
#define D_TWEET 768
#define D_USER  128
#define SLOPE   0.2f

// ---------------------------------------------------------------- utilities
__global__ void k_zero(float* p, long long n) {
  long long i = (long long)blockIdx.x * blockDim.x + threadIdx.x;
  long long stride = (long long)gridDim.x * blockDim.x;
  for (; i < n; i += stride) p[i] = 0.0f;
}

// W [K,256] f32 row-major  ->  Wt [256,K] f16 (transposed, per-column contiguous)
__global__ void k_wt_convert(const float* __restrict__ W, _Float16* __restrict__ Wt, int K) {
  int total = K * 256;
  int i = blockIdx.x * blockDim.x + threadIdx.x;
  if (i >= total) return;
  int k = i >> 8, n = i & 255;
  Wt[(long long)n * K + k] = (_Float16)W[(long long)k * 256 + n];
}

// ------------------------------------------------------------- WMMA GEMM
// H[M,256] = X[M,K] @ W[K,256], W given transposed as Wt[256][K] (f16).
// One wave computes one 16x16 tile; grid = (ceil(M/16), 4), block = 128 (4 waves).
__global__ void k_gemm_wmma(const float* __restrict__ X, const _Float16* __restrict__ Wt,
                            float* __restrict__ Hout, int M, int K) {
  const int wave = threadIdx.x >> 5;
  const int lane = threadIdx.x & 31;
  const int hf   = lane >> 4;    // lane half (0: lanes 0-15, 1: lanes 16-31)
  const int l16  = lane & 15;
  const int m0 = blockIdx.x * 16;
  const int n0 = (blockIdx.y * 4 + wave) * 16;
  int row = m0 + l16; if (row >= M) row = M - 1;   // clamp loads, mask stores

  const float*    xr = X  + (long long)row * K;
  const _Float16* wc = Wt + (long long)(n0 + l16) * K;

  v8f acc = {};
  for (int kb = 0; kb < K; kb += 32) {
    // A fragment (16x32 f16): lanes 0-15 hold K = kb+0..15, lanes 16-31 K = kb+16..31
    const int ka = kb + hf * 16;
    float4 a0 = *(const float4*)(xr + ka);
    float4 a1 = *(const float4*)(xr + ka + 4);
    float4 a2 = *(const float4*)(xr + ka + 8);
    float4 a3 = *(const float4*)(xr + ka + 12);
    v16h A;
    A[0]=(_Float16)a0.x;  A[1]=(_Float16)a0.y;  A[2]=(_Float16)a0.z;  A[3]=(_Float16)a0.w;
    A[4]=(_Float16)a1.x;  A[5]=(_Float16)a1.y;  A[6]=(_Float16)a1.z;  A[7]=(_Float16)a1.w;
    A[8]=(_Float16)a2.x;  A[9]=(_Float16)a2.y;  A[10]=(_Float16)a2.z; A[11]=(_Float16)a2.w;
    A[12]=(_Float16)a3.x; A[13]=(_Float16)a3.y; A[14]=(_Float16)a3.z; A[15]=(_Float16)a3.w;
    // B fragment (32x16 f16): lane = column n, lanes 0-15 K=kb+0..15, lanes 16-31 K=kb+16..31
    v16h B = *(const v16h*)(wc + kb + hf * 16);
    acc = __builtin_amdgcn_wmma_f32_16x16x32_f16(false, A, false, B, (short)0, acc, false, false);
  }
  // C/D layout: VGPR j -> M = hf*8 + j, N = l16
  const int mb = m0 + hf * 8;
  #pragma unroll
  for (int j = 0; j < 8; j++) {
    if (mb + j < M) Hout[(long long)(mb + j) * 256 + n0 + l16] = acc[j];
  }
}

// --------------------------------------------------- attention logits per node
// out[n][h] = sum_c H[n][h*64+c] * att[h*64+c] ; one wave per node, 8 ch per lane
__global__ void k_att_logits(const float* __restrict__ Hf, const float* __restrict__ att,
                             float* __restrict__ out, int N) {
  const int wave = threadIdx.x >> 5;
  const int lane = threadIdx.x & 31;
  const int n = blockIdx.x * (blockDim.x >> 5) + wave;
  if (n >= N) return;
  const float* hr = Hf + (long long)n * 256 + lane * 8;
  const float* ar = att + lane * 8;
  float s = 0.f;
  #pragma unroll
  for (int j = 0; j < 8; j++) s += hr[j] * ar[j];
  for (int off = 4; off >= 1; off >>= 1) s += __shfl_down(s, off, 8);
  if ((lane & 7) == 0) out[(long long)n * 4 + (lane >> 3)] = s;
}

// ------------------------------------------------------- edge softmax helpers
__device__ __forceinline__ unsigned mapf(float f) {
  unsigned u = __float_as_uint(f);
  return (u & 0x80000000u) ? ~u : (u | 0x80000000u);   // monotone map for u32 max
}
__device__ __forceinline__ float lrelu(float v) { return v > 0.f ? v : v * SLOPE; }

__global__ void k_edge_max(const int* __restrict__ src, const int* __restrict__ dst,
                           const float* __restrict__ aS, const float* __restrict__ aD,
                           unsigned* __restrict__ mm, int nE, int nLoop) {
  long long i = (long long)blockIdx.x * blockDim.x + threadIdx.x;
  long long tot = (long long)nE + nLoop;
  if (i >= tot) return;
  int s, d;
  if (i < nE) { s = src[i]; d = dst[i]; } else { s = d = (int)(i - nE); }
  const float4 as = *(const float4*)(aS + (long long)s * 4);
  const float4 ad = *(const float4*)(aD + (long long)d * 4);
  unsigned* mr = mm + (long long)d * 4;
  atomicMax(mr + 0, mapf(lrelu(as.x + ad.x)));
  atomicMax(mr + 1, mapf(lrelu(as.y + ad.y)));
  atomicMax(mr + 2, mapf(lrelu(as.z + ad.z)));
  atomicMax(mr + 3, mapf(lrelu(as.w + ad.w)));
}

// unmap in place; untouched slots (u==0, i.e. -inf) -> 0 per reference semantics
__global__ void k_finalize_max(float* m, long long n) {
  long long i = (long long)blockIdx.x * blockDim.x + threadIdx.x;
  if (i >= n) return;
  unsigned u = ((unsigned*)m)[i];
  m[i] = (u == 0u) ? 0.0f
                   : __uint_as_float((u & 0x80000000u) ? (u ^ 0x80000000u) : ~u);
}

__global__ void k_edge_expsum(const int* __restrict__ src, const int* __restrict__ dst,
                              const float* __restrict__ aS, const float* __restrict__ aD,
                              const float* __restrict__ mf,
                              float* __restrict__ p, float* __restrict__ ssum,
                              int nE, int nLoop) {
  long long i = (long long)blockIdx.x * blockDim.x + threadIdx.x;
  long long tot = (long long)nE + nLoop;
  if (i >= tot) return;
  int s, d;
  if (i < nE) { s = src[i]; d = dst[i]; } else { s = d = (int)(i - nE); }
  const float4 as = *(const float4*)(aS + (long long)s * 4);
  const float4 ad = *(const float4*)(aD + (long long)d * 4);
  const float4 mm = *(const float4*)(mf + (long long)d * 4);
  float p0 = __expf(lrelu(as.x + ad.x) - mm.x);
  float p1 = __expf(lrelu(as.y + ad.y) - mm.y);
  float p2 = __expf(lrelu(as.z + ad.z) - mm.z);
  float p3 = __expf(lrelu(as.w + ad.w) - mm.w);
  *(float4*)(p + i * 4) = make_float4(p0, p1, p2, p3);
  float* sr = ssum + (long long)d * 4;
  atomicAdd(sr + 0, p0);
  atomicAdd(sr + 1, p1);
  atomicAdd(sr + 2, p2);
  atomicAdd(sr + 3, p3);
}

// 64 threads per edge (c = 0..63), 4 heads per thread -> coalesced gather+atomics
__global__ void k_edge_scatter(const int* __restrict__ src, const int* __restrict__ dst,
                               const float* __restrict__ p, const float* __restrict__ ssum,
                               const float* __restrict__ Hs, float* __restrict__ acc,
                               int nE, int nLoop) {
  long long gid = (long long)blockIdx.x * blockDim.x + threadIdx.x;
  long long tot = ((long long)nE + nLoop) * 64;
  if (gid >= tot) return;
  long long i = gid >> 6;
  int c = (int)(gid & 63);
  int s, d;
  if (i < nE) { s = src[i]; d = dst[i]; } else { s = d = (int)(i - nE); }
  float4 pv = *(const float4*)(p + i * 4);
  float4 sv = *(const float4*)(ssum + (long long)d * 4);
  const float* hsr = Hs + (long long)s * 256 + c;
  float* ar = acc + (long long)d * 256 + c;
  atomicAdd(ar + 0,   (pv.x / (sv.x + 1e-16f)) * hsr[0]);
  atomicAdd(ar + 64,  (pv.y / (sv.y + 1e-16f)) * hsr[64]);
  atomicAdd(ar + 128, (pv.z / (sv.z + 1e-16f)) * hsr[128]);
  atomicAdd(ar + 192, (pv.w / (sv.w + 1e-16f)) * hsr[192]);
}

// out = (acc + b0 + b1 + b2) / 3  (HeteroConv mean over 3 relations, each +bias)
__global__ void k_finalize_out(float* __restrict__ out,
                               const float* __restrict__ b0, const float* __restrict__ b1,
                               const float* __restrict__ b2, long long total) {
  long long i = (long long)blockIdx.x * blockDim.x + threadIdx.x;
  if (i >= total) return;
  int c = (int)(i & 255);
  out[i] = (out[i] + b0[c] + b1[c] + b2[c]) * (1.0f / 3.0f);
}

// ---------------------------------------------------------------- host side
struct Wsp {
  float* hA; float* hB;
  float* aS; float* aD;
  unsigned* mbuf; float* sbuf; float* pbuf;
  _Float16* wt0; _Float16* wt1;
};

static void proj(const float* X, const float* W, int N, int K, _Float16* wt,
                 float* Hout, hipStream_t st) {
  int tot = K * 256;
  k_wt_convert<<<(tot + 255) / 256, 256, 0, st>>>(W, wt, K);
  dim3 g((N + 15) / 16, 4);
  k_gemm_wmma<<<g, 128, 0, st>>>(X, wt, Hout, N, K);
}

static void relation(bool homo,
                     const float* Xs, int Ns, int Ks, const float* Wsrc,
                     const float* Xd, int Nd, int Kd, const float* Wdst,
                     const float* attS, const float* attD,
                     const int* src, const int* dst, int nE,
                     float* accOut, const Wsp& w, hipStream_t st) {
  proj(Xs, Wsrc, Ns, Ks, w.wt0, w.hA, st);
  const float* hDst = w.hA;
  if (homo) {
    Nd = Ns;
  } else {
    proj(Xd, Wdst, Nd, Kd, w.wt1, w.hB, st);
    hDst = w.hB;
  }
  k_att_logits<<<(Ns + 7) / 8, 256, 0, st>>>(w.hA, attS, w.aS, Ns);
  k_att_logits<<<(Nd + 7) / 8, 256, 0, st>>>(hDst, attD, w.aD, Nd);

  int nLoop = homo ? Ns : 0;
  long long nd4 = (long long)Nd * 4;
  k_zero<<<256, 256, 0, st>>>((float*)w.mbuf, nd4);
  k_zero<<<256, 256, 0, st>>>(w.sbuf, nd4);

  long long tot = (long long)nE + nLoop;
  unsigned gEdges = (unsigned)((tot + 255) / 256);
  k_edge_max<<<gEdges, 256, 0, st>>>(src, dst, w.aS, w.aD, w.mbuf, nE, nLoop);
  k_finalize_max<<<(unsigned)((nd4 + 255) / 256), 256, 0, st>>>((float*)w.mbuf, nd4);
  k_edge_expsum<<<gEdges, 256, 0, st>>>(src, dst, w.aS, w.aD, (const float*)w.mbuf,
                                        w.pbuf, w.sbuf, nE, nLoop);
  long long tot64 = tot * 64;
  k_edge_scatter<<<(unsigned)((tot64 + 255) / 256), 256, 0, st>>>(
      src, dst, w.pbuf, w.sbuf, w.hA, accOut, nE, nLoop);
}

extern "C" void kernel_launch(void* const* d_in, const int* in_sizes, int n_in,
                              void* d_out, int out_size, void* d_ws, size_t ws_size,
                              hipStream_t stream) {
  const float* x_tweet = (const float*)d_in[0];
  const float* x_user  = (const float*)d_in[1];
  const int* src_rt = (const int*)d_in[2];  const int* dst_rt = (const int*)d_in[3];
  const int* src_rb = (const int*)d_in[4];  const int* dst_rb = (const int*)d_in[5];
  const int* src_po = (const int*)d_in[6];  const int* dst_po = (const int*)d_in[7];
  const int* src_pb = (const int*)d_in[8];  const int* dst_pb = (const int*)d_in[9];
  const int* src_iw = (const int*)d_in[10]; const int* dst_iw = (const int*)d_in[11];
  const int* src_ib = (const int*)d_in[12]; const int* dst_ib = (const int*)d_in[13];
  const float* W_rt = (const float*)d_in[14]; const float* as_rt = (const float*)d_in[15];
  const float* ad_rt = (const float*)d_in[16]; const float* b_rt = (const float*)d_in[17];
  const float* W_rb = (const float*)d_in[18]; const float* as_rb = (const float*)d_in[19];
  const float* ad_rb = (const float*)d_in[20]; const float* b_rb = (const float*)d_in[21];
  const float* W_iw = (const float*)d_in[22]; const float* as_iw = (const float*)d_in[23];
  const float* ad_iw = (const float*)d_in[24]; const float* b_iw = (const float*)d_in[25];
  const float* W_ib = (const float*)d_in[26]; const float* as_ib = (const float*)d_in[27];
  const float* ad_ib = (const float*)d_in[28]; const float* b_ib = (const float*)d_in[29];
  const float* Ws_po = (const float*)d_in[30]; const float* Wd_po = (const float*)d_in[31];
  const float* as_po = (const float*)d_in[32]; const float* ad_po = (const float*)d_in[33];
  const float* b_po = (const float*)d_in[34];
  const float* Ws_pb = (const float*)d_in[35]; const float* Wd_pb = (const float*)d_in[36];
  const float* as_pb = (const float*)d_in[37]; const float* ad_pb = (const float*)d_in[38];
  const float* b_pb = (const float*)d_in[39];

  const int E = in_sizes[2];

  // workspace carve-out (256B aligned), ~220 MB total
  char* base = (char*)d_ws;
  auto carve = [&](size_t bytes) -> char* {
    char* q = base;
    base += (bytes + 255) & ~(size_t)255;
    return q;
  };
  Wsp w;
  w.hA   = (float*)carve((size_t)N_TWEET * 256 * 4);
  w.hB   = (float*)carve((size_t)N_TWEET * 256 * 4);
  w.aS   = (float*)carve((size_t)N_TWEET * 4 * 4);
  w.aD   = (float*)carve((size_t)N_TWEET * 4 * 4);
  w.mbuf = (unsigned*)carve((size_t)N_TWEET * 4 * 4);
  w.sbuf = (float*)carve((size_t)N_TWEET * 4 * 4);
  w.pbuf = (float*)carve(((size_t)E + N_TWEET) * 4 * 4);
  w.wt0  = (_Float16*)carve((size_t)D_TWEET * 256 * 2);
  w.wt1  = (_Float16*)carve((size_t)D_TWEET * 256 * 2);

  float* out  = (float*)d_out;
  float* outT = out;                                // [NT,256]
  float* outU = out + (long long)N_TWEET * 256;     // [NU,256]

  long long totOut = (long long)(N_TWEET + N_USER) * 256;
  k_zero<<<1024, 256, 0, stream>>>(out, totOut);

  // tweet-destination relations -> accumulate into outT
  relation(true,  x_tweet, N_TWEET, D_TWEET, W_rt, nullptr, 0, 0, nullptr,
           as_rt, ad_rt, src_rt, dst_rt, E, outT, w, stream);
  relation(true,  x_tweet, N_TWEET, D_TWEET, W_rb, nullptr, 0, 0, nullptr,
           as_rb, ad_rb, src_rb, dst_rb, E, outT, w, stream);
  relation(false, x_user,  N_USER,  D_USER,  Ws_po, x_tweet, N_TWEET, D_TWEET, Wd_po,
           as_po, ad_po, src_po, dst_po, E, outT, w, stream);
  // user-destination relations -> accumulate into outU
  relation(false, x_tweet, N_TWEET, D_TWEET, Ws_pb, x_user, N_USER, D_USER, Wd_pb,
           as_pb, ad_pb, src_pb, dst_pb, E, outU, w, stream);
  relation(true,  x_user,  N_USER,  D_USER,  W_iw, nullptr, 0, 0, nullptr,
           as_iw, ad_iw, src_iw, dst_iw, E, outU, w, stream);
  relation(true,  x_user,  N_USER,  D_USER,  W_ib, nullptr, 0, 0, nullptr,
           as_ib, ad_ib, src_ib, dst_ib, E, outU, w, stream);

  long long totT = (long long)N_TWEET * 256;
  long long totU = (long long)N_USER * 256;
  k_finalize_out<<<(unsigned)((totT + 255) / 256), 256, 0, stream>>>(outT, b_rt, b_rb, b_po, totT);
  k_finalize_out<<<(unsigned)((totU + 255) / 256), 256, 0, stream>>>(outU, b_pb, b_iw, b_ib, totU);
}
